// HybridQLSTM_65481071395454
// MI455X (gfx1250) — matrix operand order repaired and checked by reading, not compiled, not run on hardware
//
#include <hip/hip_runtime.h>
#include <hip/hip_bf16.h>

// Problem constants (reference: T=1024, B=64, D=512, H=512)
#define T_STEPS 1024
#define BATCH   64
#define DIM     512
#define HID     512
#define DH      1024            // D + H (GEMM contraction length)
#define NTILE   16              // hidden units per workgroup (one WMMA N tile)
#define NWG     (HID / NTILE)   // 32 persistent workgroups
#define THREADS 128             // 4 waves (wave32): one gate per wave
#define LPAD    16              // bf16 elements of row padding (32B) for LDS bank spread
#define LSTR    (DH + LPAD)     // padded LDS row stride in bf16 elements

typedef __attribute__((ext_vector_type(16))) __bf16 v16bf;
typedef __attribute__((ext_vector_type(8)))  __bf16 v8bf;
typedef __attribute__((ext_vector_type(8)))  float  v8f;

__device__ __forceinline__ unsigned short f32_to_bf16_rne(float f) {
    unsigned u = __float_as_uint(f);
    unsigned r = 0x7FFFu + ((u >> 16) & 1u);
    return (unsigned short)((u + r) >> 16);
}

__device__ __forceinline__ float sigf(float x) {
    return 1.0f / (1.0f + __expf(-x));
}

// CDNA5 async global->LDS copy, 16 bytes per lane. VDST VGPR carries the LDS
// byte offset (generic LDS pointers truncate to the LDS offset); VADDR is the
// 64-bit global address. Tracked with ASYNCcnt.
__device__ __forceinline__ void async_copy_b128(const void* gptr, void* lptr) {
    unsigned lds = (unsigned)(unsigned long long)(uintptr_t)lptr;
    asm volatile("global_load_async_to_lds_b128 %0, %1, off"
                 :: "v"(lds), "v"((unsigned long long)(uintptr_t)gptr)
                 : "memory");
}

__device__ __forceinline__ void wait_async0() {
    asm volatile("s_wait_asynccnt 0x0" ::: "memory");
}

// ---------------------------------------------------------------------------
// Init: zero grid-barrier counter and both h double-buffers (h0 = 0).
// Must run every call: harness poisons workspace with 0xAA.
// ---------------------------------------------------------------------------
__global__ void qlstm_init_kernel(unsigned* __restrict__ bar,
                                  unsigned short* __restrict__ hbf) {
    int idx = blockIdx.x * blockDim.x + threadIdx.x;
    if (idx == 0) *bar = 0u;
    if (idx < 2 * BATCH * HID) hbf[idx] = (unsigned short)0;
}

// ---------------------------------------------------------------------------
// Convert the four gate weight matrices (each [H, D+H] f32) into one bf16
// matrix Wbf[4H, D+H], row = gate*H + r (matches reference concat order).
// ---------------------------------------------------------------------------
__global__ void qlstm_wconv_kernel(const float* __restrict__ Wf,
                                   const float* __restrict__ Wi,
                                   const float* __restrict__ Wg,
                                   const float* __restrict__ Wo,
                                   unsigned short* __restrict__ Wbf) {
    int idx = blockIdx.x * blockDim.x + threadIdx.x;
    if (idx >= 4 * HID * DH) return;
    int row = idx >> 10;            // / DH
    int k   = idx & (DH - 1);
    int g   = row >> 9;             // / HID
    int r   = row & (HID - 1);
    const float* W = (g == 0) ? Wf : (g == 1) ? Wi : (g == 2) ? Wg : Wo;
    Wbf[idx] = f32_to_bf16_rne(W[(size_t)r * DH + k]);
}

// ---------------------------------------------------------------------------
// One-time f32 -> bf16 conversion of the whole input tensor [T,B,D].
// Memory-bound pre-pass (~200MB total traffic, ~9us at 23.3 TB/s); removes
// all conversion work from the serialized recurrent loop.
// ---------------------------------------------------------------------------
__global__ void qlstm_xconv_kernel(const float* __restrict__ x,
                                   unsigned short* __restrict__ xbf) {
    int idx = blockIdx.x * blockDim.x + threadIdx.x;   // float4 group index
    if (idx >= T_STEPS * BATCH * DIM / 4) return;
    float4 v = *(const float4*)(x + (size_t)idx * 4);
    ushort4 p;
    p.x = f32_to_bf16_rne(v.x); p.y = f32_to_bf16_rne(v.y);
    p.z = f32_to_bf16_rne(v.z); p.w = f32_to_bf16_rne(v.w);
    *(ushort4*)(xbf + (size_t)idx * 4) = p;
}

// ---------------------------------------------------------------------------
// Persistent recurrent kernel. Grid = 32 WGs; WG b owns hidden units
// [16b, 16b+16). LDS holds: comb = [bf16(x_t) | h] (64x1024), the WG's W
// slice (4 gates x 16 rows x 1024), the f32 gate-exchange tile, and the
// persistent c state. Per step: async-stage comb -> WMMA GEMM -> gate math ->
// publish h slice -> device-scope atomic barrier.
// ---------------------------------------------------------------------------
__launch_bounds__(THREADS, 1)
__global__ void qlstm_main_kernel(const unsigned short* __restrict__ xbf,  // [T,B,D] bf16
                                  const unsigned short* __restrict__ Wbf,  // [4H, DH] bf16
                                  const float* __restrict__ bfv,
                                  const float* __restrict__ biv,
                                  const float* __restrict__ bgv,
                                  const float* __restrict__ bov,
                                  const float* __restrict__ q_scale,
                                  const float* __restrict__ q_bias,
                                  float* __restrict__ out,                 // [T*B*H + 2*B*H]
                                  unsigned short* __restrict__ hbuf,       // [2][B*H] bf16
                                  unsigned* __restrict__ bar) {
    const int tid  = threadIdx.x;
    const int wave = tid >> 5;        // gate id: 0=f 1=i 2=g 3=o
    const int lane = tid & 31;
    const int half = lane >> 4;       // 0: lanes 0-15, 1: lanes 16-31
    const int lrow = lane & 15;
    const int n0   = blockIdx.x * NTILE;

    extern __shared__ __align__(32) char smem[];
    unsigned short* comb = (unsigned short*)smem;          // [64][LSTR]
    unsigned short* wl   = comb + BATCH * LSTR;            // [64 rows = 4 gates x 16][LSTR]
    float* zex = (float*)(wl + 64 * LSTR);                 // [4][64][16]
    float* cst = zex + 4 * BATCH * NTILE;                  // [64][16] persistent c
    float* qsl = cst + BATCH * NTILE;                      // [16]
    float* qbl = qsl + NTILE;                              // [16]

    // ---- one-time: pull this WG's W slice into LDS (stays resident) ----
    for (int i = tid; i < 64 * (DH / 8); i += THREADS) {   // 8192 x 16B async copies
        int gj = i >> 7;                                   // 0..63 = gate*16 + j
        int k8 = (i & 127) * 8;
        int g  = gj >> 4, j = gj & 15;
        async_copy_b128(Wbf + (size_t)(g * HID + n0 + j) * DH + k8,
                        wl + (size_t)gj * LSTR + k8);
    }
    if (tid < NTILE) { qsl[tid] = q_scale[n0 + tid]; qbl[tid] = q_bias[n0 + tid]; }
    for (int i = tid; i < BATCH * NTILE; i += THREADS) cst[i] = 0.0f;

    // bias for this wave's gate at output column (lane&15)
    const float* bptr = (wave == 0) ? bfv : (wave == 1) ? biv : (wave == 2) ? bgv : bov;
    const float bias  = bptr[n0 + lrow];

    const unsigned short* wg = wl + (size_t)wave * NTILE * LSTR;

    for (int t = 0; t < T_STEPS; ++t) {
        const unsigned short* hrd = hbuf + (size_t)(t & 1) * BATCH * HID;
        unsigned short*       hwr = hbuf + (size_t)((t + 1) & 1) * BATCH * HID;

        // ---- async-stage comb = [ bf16(x_t) | h_{t-1} ] into LDS ----
        {
            const unsigned short* xt = xbf + (size_t)t * BATCH * DIM;
            for (int i = tid; i < BATCH * DIM / 8; i += THREADS) {   // 4096 x 16B
                int b = i >> 6, k8 = (i & 63) * 8;
                async_copy_b128(xt + (size_t)b * DIM + k8,
                                comb + (size_t)b * LSTR + k8);
            }
            for (int i = tid; i < BATCH * HID / 8; i += THREADS) {   // 4096 x 16B
                int b = i >> 6, k8 = (i & 63) * 8;
                async_copy_b128(hrd + (size_t)b * HID + k8,
                                comb + (size_t)b * LSTR + DIM + k8);
            }
            if (t + 1 < T_STEPS) {                                   // warm L2 for x_{t+1}
                const unsigned short* xn = xbf + (size_t)(t + 1) * BATCH * DIM;
                for (int i = tid; i < BATCH * DIM / 8; i += THREADS * 16)
                    __builtin_prefetch(xn + (size_t)(i >> 6) * DIM + (i & 63) * 8, 0, 1);
            }
        }
        wait_async0();       // drain ASYNCcnt before cross-wave LDS visibility
        __syncthreads();

        // ---- GEMM: z_gate[64x16] = comb(64x1024) @ W_gate^T(1024x16) + b ----
        v8f acc[4];
        #pragma unroll
        for (int m = 0; m < 4; ++m)
            #pragma unroll
            for (int v = 0; v < 8; ++v) acc[m][v] = bias;

        for (int kk = 0; kk < DH; kk += 32) {
            // B tile 32x16: lane = column (lane&15), 16 contiguous K
            const v16bf bmat = *(const v16bf*)(wg + (size_t)lrow * LSTR + kk + half * 16);
            #pragma unroll
            for (int m = 0; m < 4; ++m) {
                // A tile 16x32: lane row, 16 contiguous K (0..15 / 16..31)
                const int r = m * 16 + lrow;
                v8bf lo = *(const v8bf*)(comb + (size_t)r * LSTR + kk + half * 16);
                v8bf hi = *(const v8bf*)(comb + (size_t)r * LSTR + kk + half * 16 + 8);
                v16bf amat;
                #pragma unroll
                for (int e = 0; e < 8; ++e) { amat[e] = lo[e]; amat[e + 8] = hi[e]; }
                acc[m] = __builtin_amdgcn_wmma_f32_16x16x32_bf16(
                    false, amat, false, bmat, (short)0, acc[m], false, false);
            }
        }

        // ---- exchange gate tiles through LDS (C layout: M=v+8*half, N=lane&15)
        {
            float* zg = zex + (size_t)wave * BATCH * NTILE;
            #pragma unroll
            for (int m = 0; m < 4; ++m)
                #pragma unroll
                for (int v = 0; v < 8; ++v)
                    zg[(m * 16 + v + 8 * half) * NTILE + lrow] = acc[m][v];
        }
        __syncthreads();

        // ---- gate math + recurrence (64x16 elements over 128 threads) ----
        for (int i = tid; i < BATCH * NTILE; i += THREADS) {
            int b = i >> 4, j = i & 15;
            float qs = qsl[j], qb = qbl[j];
            float fg = tanhf(sigf(zex[0 * BATCH * NTILE + i]) * qs + qb);
            float ig = tanhf(sigf(zex[1 * BATCH * NTILE + i]) * qs + qb);
            float gg = tanhf(tanhf(zex[2 * BATCH * NTILE + i]) * qs + qb);
            float og = tanhf(sigf(zex[3 * BATCH * NTILE + i]) * qs + qb);
            float c = fg * cst[i] + ig * gg;
            cst[i] = c;
            float h = og * tanhf(c);
            out[(size_t)t * BATCH * HID + (size_t)b * HID + n0 + j] = h;
            hwr[(size_t)b * HID + n0 + j] = f32_to_bf16_rne(h);
            if (t == T_STEPS - 1) {
                out[(size_t)T_STEPS * BATCH * HID + (size_t)b * HID + n0 + j] = h;   // hx
                out[(size_t)T_STEPS * BATCH * HID + BATCH * HID
                    + (size_t)b * HID + n0 + j] = c;                                 // cx
            }
        }

        // ---- device-scope grid barrier (monotonic counter, graph-safe) ----
        __threadfence();
        __syncthreads();
        if (tid == 0) {
            atomicAdd(bar, 1u);
            const unsigned target = (unsigned)NWG * (unsigned)(t + 1);
            while (__hip_atomic_load(bar, __ATOMIC_ACQUIRE, __HIP_MEMORY_SCOPE_AGENT) < target)
                __builtin_amdgcn_s_sleep(8);
        }
        __syncthreads();
        __threadfence();
    }
}

// ---------------------------------------------------------------------------
extern "C" void kernel_launch(void* const* d_in, const int* in_sizes, int n_in,
                              void* d_out, int out_size, void* d_ws, size_t ws_size,
                              hipStream_t stream) {
    (void)in_sizes; (void)n_in; (void)out_size; (void)ws_size;
    const float* x   = (const float*)d_in[0];
    const float* Wf  = (const float*)d_in[1];
    const float* bfv = (const float*)d_in[2];
    const float* Wi  = (const float*)d_in[3];
    const float* biv = (const float*)d_in[4];
    const float* Wg  = (const float*)d_in[5];
    const float* bgv = (const float*)d_in[6];
    const float* Wo  = (const float*)d_in[7];
    const float* bov = (const float*)d_in[8];
    const float* qs  = (const float*)d_in[9];
    const float* qb  = (const float*)d_in[10];
    float* out = (float*)d_out;

    // Workspace carve: Wbf (4MB) | xbf (64MB bf16) | h double buffer | barrier
    unsigned short* Wbf  = (unsigned short*)d_ws;
    unsigned short* xbf  = Wbf + (size_t)4 * HID * DH;
    unsigned short* hbuf = xbf + (size_t)T_STEPS * BATCH * DIM;
    unsigned*       bar  = (unsigned*)(hbuf + (size_t)2 * BATCH * HID);

    qlstm_init_kernel<<<(2 * BATCH * HID + 255) / 256, 256, 0, stream>>>(bar, hbuf);
    qlstm_wconv_kernel<<<(4 * HID * DH + 255) / 256, 256, 0, stream>>>(Wf, Wi, Wg, Wo, Wbf);
    qlstm_xconv_kernel<<<(T_STEPS * BATCH * DIM / 4 + 255) / 256, 256, 0, stream>>>(x, xbf);

    // Dynamic LDS: comb + W slice (padded rows) + gate exchange + c + q params
    const size_t smem = (size_t)(BATCH + 64) * LSTR * sizeof(unsigned short)
                      + (size_t)4 * BATCH * NTILE * sizeof(float)
                      + (size_t)BATCH * NTILE * sizeof(float)
                      + (size_t)2 * NTILE * sizeof(float);
    hipFuncSetAttribute(reinterpret_cast<const void*>(&qlstm_main_kernel),
                        hipFuncAttributeMaxDynamicSharedMemorySize, (int)smem);
    qlstm_main_kernel<<<NWG, THREADS, smem, stream>>>(
        xbf, Wbf, bfv, biv, bgv, bov, qs, qb, out, hbuf, bar);
}
